// GAT1_6605659701636
// MI455X (gfx1250) — matrix operand order repaired and checked by reading, compiled, not run on hardware
//
#include <hip/hip_runtime.h>
#include <hip/hip_bf16.h>

// ---------------------------------------------------------------------------
// Types for CDNA5 WMMA (wave32, 16x16x32 bf16 -> f32)
// ---------------------------------------------------------------------------
typedef __attribute__((ext_vector_type(16))) __bf16 v16bf;
typedef __attribute__((ext_vector_type(8)))  float  v8f;

#define NEG_SLOPE 0.2f
#define BN_EPS    1e-5f

// ---------------------------------------------------------------------------
// Helpers
// ---------------------------------------------------------------------------
__device__ __forceinline__ unsigned short f2bf(float x) {
  // round-to-nearest-even fp32 -> bf16
  unsigned u = __float_as_uint(x);
  u = u + 0x7FFFu + ((u >> 16) & 1u);
  return (unsigned short)(u >> 16);
}

__device__ __forceinline__ float elu1(float v) {
  return v > 0.f ? v : (__expf(v) - 1.f);
}

__device__ __forceinline__ float atomicMaxFloat(float* addr, float value) {
  if (__float_as_int(value) >= 0) {
    return __int_as_float(atomicMax((int*)addr, __float_as_int(value)));
  } else {
    return __uint_as_float(atomicMin((unsigned*)addr, (unsigned)__float_as_int(value)));
  }
}

// ---------------------------------------------------------------------------
// fp32 -> bf16 conversion with zero padding (A matrices, row-major)
// ---------------------------------------------------------------------------
__global__ void k_cvt_bf16(const float* __restrict__ src, unsigned short* __restrict__ dst,
                           long long srcR, int srcC, long long dstR, int dstC) {
  long long total = dstR * (long long)dstC;
  long long stride = (long long)gridDim.x * blockDim.x;
  for (long long i = blockIdx.x * (long long)blockDim.x + threadIdx.x; i < total; i += stride) {
    long long r = i / dstC;
    int c = (int)(i - r * (long long)dstC);
    float v = (r < srcR && c < srcC) ? src[r * (long long)srcC + c] : 0.f;
    dst[i] = f2bf(v);
  }
}

// fp32 [srcR, srcC] -> bf16 TRANSPOSED [srcC, dstRp] with zero pad on K (rows)
__global__ void k_cvt_bf16_T(const float* __restrict__ src, unsigned short* __restrict__ dst,
                             long long srcR, int srcC, long long dstRp) {
  long long total = (long long)srcC * dstRp;
  long long stride = (long long)gridDim.x * blockDim.x;
  for (long long i = blockIdx.x * (long long)blockDim.x + threadIdx.x; i < total; i += stride) {
    long long c = i / dstRp;           // output row = source column (N dim)
    long long r = i - c * dstRp;       // output col = source row    (K dim)
    float v = (r < srcR) ? src[r * (long long)srcC + c] : 0.f;
    dst[i] = f2bf(v);
  }
}

__global__ void k_fill(float* __restrict__ p, float v, long long n) {
  long long stride = (long long)gridDim.x * blockDim.x;
  for (long long i = blockIdx.x * (long long)blockDim.x + threadIdx.x; i < n; i += stride)
    p[i] = v;
}

// ---------------------------------------------------------------------------
// Tiled WMMA GEMM: C[M,256] = A[M,K](bf16) @ B[K,256](bf16) (+bias), f32 out.
// B passed TRANSPOSED: Bt[256][K].  Block tile 128x256, K-step 32, 8 waves.
// Wave w owns rows [16w,16w+16) x all 256 cols: 1 A fragment -> 16 WMMAs/K-step.
// Global->LDS staging is software-pipelined: next tile's global loads are
// issued (into 6 distinct uint4 regs) right after the first barrier so they
// overlap the 16 WMMAs; they are stored to LDS at the top of the next step.
// ---------------------------------------------------------------------------
#define GT_M 128
#define GT_K 32

__global__ __launch_bounds__(256) void k_gemm_bf16(
    const unsigned short* __restrict__ A,   // M x K (bf16 bits), K % 32 == 0
    const unsigned short* __restrict__ Bt,  // 256 x K (bf16 bits), N-major
    float* __restrict__ C,                  // M x 256
    const float* __restrict__ bias,         // 256 or nullptr
    int M, int K) {
  __shared__ unsigned short As[GT_M][GT_K];   // 8 KB
  __shared__ unsigned short Bs[256][GT_K];    // 16 KB, [n][k], 64B rows

  const int tid  = threadIdx.x;
  const int lane = tid & 31;
  const int wave = tid >> 5;
  const int bm = blockIdx.x * GT_M;

  v8f acc[16];
#pragma unroll
  for (int i = 0; i < 16; ++i) acc[i] = (v8f){};

  // staging coordinates (per thread)
  const int sr = tid >> 2;             // A rows: sr and sr+64
  const int sc = (tid & 3) * 8;        // A col block (8 bf16 = 16B)
  const bool in0 = (bm + sr)      < M;
  const bool in1 = (bm + 64 + sr) < M;
  const unsigned short* a0p = A + (size_t)(bm + sr)      * K + sc;
  const unsigned short* a1p = A + (size_t)(bm + 64 + sr) * K + sc;
  const unsigned short* bgp = Bt + (size_t)tid * K;

  // fragment addressing (16-bit matrix 16x32 lane layout, wave32)
  const int fm   = lane & 15;
  const int koff = (lane >> 4) * 8;    // lanes 16-31 hold K+8 block
  const int kb16 = (lane >> 4) * 16;   // lanes 16-31 hold K 16..31

  uint4 ra0, ra1, rb0, rb1, rb2, rb3;  // in-flight staging registers
  {
    const uint4 z = make_uint4(0u, 0u, 0u, 0u);
    ra0 = in0 ? *(const uint4*)(a0p) : z;
    ra1 = in1 ? *(const uint4*)(a1p) : z;
    const uint4* bg = (const uint4*)(bgp);
    rb0 = bg[0]; rb1 = bg[1]; rb2 = bg[2]; rb3 = bg[3];
  }

  for (int k0 = 0; k0 < K; k0 += GT_K) {
    // ---- commit staged regs to LDS ----
    *(uint4*)(&As[sr][sc])      = ra0;
    *(uint4*)(&As[64 + sr][sc]) = ra1;
    {
      uint4* bs = (uint4*)(&Bs[tid][0]);
      bs[0] = rb0; bs[1] = rb1; bs[2] = rb2; bs[3] = rb3;
    }
    __syncthreads();

    // ---- prefetch next tile (overlaps the 16 WMMAs below) ----
    if (k0 + GT_K < K) {
      const uint4 z = make_uint4(0u, 0u, 0u, 0u);
      int kn = k0 + GT_K;
      ra0 = in0 ? *(const uint4*)(a0p + kn) : z;
      ra1 = in1 ? *(const uint4*)(a1p + kn) : z;
      const uint4* bg = (const uint4*)(bgp + kn);
      rb0 = bg[0]; rb1 = bg[1]; rb2 = bg[2]; rb3 = bg[3];
    }

    // ---- A fragment for this wave's 16-row stripe (2x b128 DS loads) ----
    union { uint4 u[2]; v16bf v; } af;
    af.u[0] = *(const uint4*)(&As[wave * 16 + fm][koff]);
    af.u[1] = *(const uint4*)(&As[wave * 16 + fm][16 + koff]);

    // ---- 16 N-subtiles: B fragment + WMMA ----
#pragma unroll
    for (int nt = 0; nt < 16; ++nt) {
      union { uint4 u[2]; v16bf v; } bf;
      const unsigned short* bp = &Bs[(lane & 15) + nt * 16][koff];
      bf.u[0] = *(const uint4*)(bp);
      bf.u[1] = *(const uint4*)(bp + 16);
      acc[nt] = __builtin_amdgcn_wmma_f32_16x16x32_bf16(
          /*neg_a=*/false, af.v, /*neg_b=*/false, bf.v,
          /*c_mod=*/(short)0, acc[nt], /*reuse_a=*/false, /*reuse_b=*/false);
    }
    __syncthreads();
  }

  // ---- epilogue: C/D layout — VGPR r: lanes 0-15 -> M=r, lanes 16-31 -> M=r+8
  const int rbase = bm + wave * 16 + (lane >> 4) * 8;
  const int col0  = lane & 15;
#pragma unroll
  for (int nt = 0; nt < 16; ++nt) {
    int col = col0 + nt * 16;
    float badd = bias ? bias[col] : 0.f;
#pragma unroll
    for (int r = 0; r < 8; ++r) {
      int row = rbase + r;
      if (row < M) C[(size_t)row * 256 + col] = acc[nt][r] + badd;
    }
  }
}

// ---------------------------------------------------------------------------
// Per-node attention dots: s_src[n] = h[n].a_src, s_dst[n] = h[n].a_dst
// ---------------------------------------------------------------------------
__global__ void k_dots(const float* __restrict__ h,
                       const float* __restrict__ a_src, const float* __restrict__ a_dst,
                       float* __restrict__ s_src, float* __restrict__ s_dst, int N) {
  int wid  = (int)((blockIdx.x * (long long)blockDim.x + threadIdx.x) >> 5);
  int lane = threadIdx.x & 31;
  if (wid >= N) return;
  const float* row = h + (size_t)wid * 256;
  float ss = 0.f, sd = 0.f;
#pragma unroll
  for (int j = lane; j < 256; j += 32) {
    float v = row[j];
    ss += v * a_src[j];
    sd += v * a_dst[j];
  }
#pragma unroll
  for (int off = 16; off; off >>= 1) {
    ss += __shfl_down(ss, off, 32);
    sd += __shfl_down(sd, off, 32);
  }
  if (lane == 0) { s_src[wid] = ss; s_dst[wid] = sd; }
}

// ---------------------------------------------------------------------------
// Edge phase 1: e = leaky_relu(s_src[src]+s_dst[dst]); segment-max into m[dst]
// ---------------------------------------------------------------------------
__global__ void k_edge1(const int* __restrict__ src, const int* __restrict__ dst,
                        long long E, long long ET,
                        const float* __restrict__ ssrc, const float* __restrict__ sdst,
                        float* __restrict__ ebuf, float* __restrict__ m) {
  long long stride = (long long)gridDim.x * blockDim.x;
  for (long long t = blockIdx.x * (long long)blockDim.x + threadIdx.x; t < ET; t += stride) {
    int s = t < E ? src[t] : (int)(t - E);
    int d = t < E ? dst[t] : (int)(t - E);
    float e = ssrc[s] + sdst[d];
    e = e > 0.f ? e : NEG_SLOPE * e;
    ebuf[t] = e;
    atomicMaxFloat(m + d, e);
  }
}

// Edge phase 2: ex = exp(e - m[dst]); segment-sum into denom[dst]
__global__ void k_edge2(const int* __restrict__ src, const int* __restrict__ dst,
                        long long E, long long ET,
                        float* __restrict__ ebuf,
                        const float* __restrict__ m, float* __restrict__ den) {
  long long stride = (long long)gridDim.x * blockDim.x;
  for (long long t = blockIdx.x * (long long)blockDim.x + threadIdx.x; t < ET; t += stride) {
    int d = t < E ? dst[t] : (int)(t - E);
    float ex = __expf(ebuf[t] - m[d]);
    ebuf[t] = ex;
    atomicAdd(den + d, ex);
  }
}

// Edge phase 3: acc[dst] += alpha * h[src].  Block per edge, 256 feature lanes.
__global__ void k_edge3(const int* __restrict__ src, const int* __restrict__ dst,
                        long long E, long long ET,
                        const float* __restrict__ ebuf, const float* __restrict__ den,
                        const float* __restrict__ h, float* __restrict__ acc) {
  int f = threadIdx.x;
  for (long long t = blockIdx.x; t < ET; t += gridDim.x) {
    int s = t < E ? src[t] : (int)(t - E);
    int d = t < E ? dst[t] : (int)(t - E);
    float alpha = ebuf[t] / den[d];
    atomicAdd(acc + (size_t)d * 256 + f, h[(size_t)s * 256 + f] * alpha);
  }
}

// acc = elu(acc + bias); also emit bf16 copy for the next WMMA GEMM
__global__ void k_bias_elu(float* __restrict__ acc, const float* __restrict__ bias,
                           unsigned short* __restrict__ outb, long long total) {
  long long stride = (long long)gridDim.x * blockDim.x;
  for (long long i = blockIdx.x * (long long)blockDim.x + threadIdx.x; i < total; i += stride) {
    int f = (int)(i & 255);
    float v = elu1(acc[i] + bias[f]);
    acc[i] = v;
    outb[i] = f2bf(v);
  }
}

// BatchNorm stats: one block per feature (biased variance)
__global__ void k_bn_stats(const float* __restrict__ X, float* __restrict__ mu,
                           float* __restrict__ var, int N) {
  __shared__ float sh[256], sh2[256];
  int f = blockIdx.x, tid = threadIdx.x;
  float s = 0.f, s2 = 0.f;
  for (int r = tid; r < N; r += blockDim.x) {
    float v = X[(size_t)r * 256 + f];
    s += v; s2 += v * v;
  }
  sh[tid] = s; sh2[tid] = s2;
  __syncthreads();
  for (int st = 128; st; st >>= 1) {
    if (tid < st) { sh[tid] += sh[tid + st]; sh2[tid] += sh2[tid + st]; }
    __syncthreads();
  }
  if (tid == 0) {
    float m = sh[0] / (float)N;
    mu[f] = m;
    var[f] = sh2[0] / (float)N - m * m;
  }
}

// y = elu(gamma*(x-mu)*rsqrt(var+eps)+beta); optional bf16 side output
__global__ void k_bn_apply(const float* __restrict__ X, const float* __restrict__ mu,
                           const float* __restrict__ var, const float* __restrict__ g,
                           const float* __restrict__ be, float* __restrict__ outf,
                           unsigned short* __restrict__ outb, long long total) {
  long long stride = (long long)gridDim.x * blockDim.x;
  for (long long i = blockIdx.x * (long long)blockDim.x + threadIdx.x; i < total; i += stride) {
    int f = (int)(i & 255);
    float v = g[f] * (X[i] - mu[f]) * rsqrtf(var[f] + BN_EPS) + be[f];
    v = elu1(v);
    outf[i] = v;
    if (outb) outb[i] = f2bf(v);
  }
}

// ---------------------------------------------------------------------------
// Orchestration
// ---------------------------------------------------------------------------
extern "C" void kernel_launch(void* const* d_in, const int* in_sizes, int n_in,
                              void* d_out, int out_size, void* d_ws, size_t ws_size,
                              hipStream_t stream) {
  const float* x    = (const float*)d_in[0];
  const int*  edges = (const int*)d_in[1];
  const float* W1   = (const float*)d_in[2];
  const float* a1s  = (const float*)d_in[3];
  const float* a1d  = (const float*)d_in[4];
  const float* b1   = (const float*)d_in[5];
  const float* W2   = (const float*)d_in[6];
  const float* a2s  = (const float*)d_in[7];
  const float* a2d  = (const float*)d_in[8];
  const float* b2   = (const float*)d_in[9];
  const float* lw1  = (const float*)d_in[10];
  const float* lb1  = (const float*)d_in[11];
  const float* g1   = (const float*)d_in[12];
  const float* be1  = (const float*)d_in[13];
  const float* lw2  = (const float*)d_in[14];
  const float* lb2  = (const float*)d_in[15];
  const float* g2   = (const float*)d_in[16];
  const float* be2  = (const float*)d_in[17];

  const int F = in_sizes[3];                         // 256
  const int D = in_sizes[2] / F;                     // 2613
  const long long N = (long long)in_sizes[0] / D;    // 50000
  const long long E = (long long)in_sizes[1] / 2;    // 800000
  const int Kp1 = (D + 31) & ~31;                    // 2624
  const int* srcIdx = edges;
  const int* dstIdx = edges + E;
  const long long ET = E + N;
  const long long NF = N * (long long)F;

  // workspace carve-out
  char* p = (char*)d_ws;
  auto alloc = [&](size_t b) { void* r = (void*)p; p += (b + 255) & ~(size_t)255; return r; };
  unsigned short* xb  = (unsigned short*)alloc((size_t)N * Kp1 * 2);
  unsigned short* w1t = (unsigned short*)alloc((size_t)F * Kp1 * 2);  // W1^T [256][Kp1]
  unsigned short* wt  = (unsigned short*)alloc((size_t)F * F * 2);    // reused [256][256]^T
  float* hA  = (float*)alloc((size_t)NF * 4);   // GEMM outputs / gather source
  float* hB  = (float*)alloc((size_t)NF * 4);   // aggregation accumulator
  unsigned short* hb = (unsigned short*)alloc((size_t)NF * 2); // bf16 GEMM input
  float* ssrc = (float*)alloc((size_t)N * 4);
  float* sdst = (float*)alloc((size_t)N * 4);
  float* mbuf = (float*)alloc((size_t)N * 4);
  float* den  = (float*)alloc((size_t)N * 4);
  float* ebuf = (float*)alloc((size_t)ET * 4);
  float* mu   = (float*)alloc((size_t)F * 4);
  float* var  = (float*)alloc((size_t)F * 4);

  auto gsz = [](long long total, int tpb) {
    long long b = (total + tpb - 1) / tpb;
    if (b > (1ll << 20)) b = 1ll << 20;
    if (b < 1) b = 1;
    return (unsigned)b;
  };

  const dim3 ggrid((unsigned)((N + GT_M - 1) / GT_M), 1);

  // GAT layer: input features in hA (fp32), output elu(agg + bias) -> hB + hb
  auto gat_layer = [&](const float* aS, const float* aD, const float* bias) {
    k_dots<<<gsz(N * 32, 256), 256, 0, stream>>>(hA, aS, aD, ssrc, sdst, (int)N);
    k_fill<<<gsz(N, 256), 256, 0, stream>>>(mbuf, -INFINITY, N);
    k_fill<<<gsz(N, 256), 256, 0, stream>>>(den, 0.f, N);
    k_fill<<<gsz(NF, 256), 256, 0, stream>>>(hB, 0.f, NF);
    k_edge1<<<gsz(ET, 256), 256, 0, stream>>>(srcIdx, dstIdx, E, ET, ssrc, sdst, ebuf, mbuf);
    k_edge2<<<gsz(ET, 256), 256, 0, stream>>>(srcIdx, dstIdx, E, ET, ebuf, mbuf, den);
    unsigned gb = (unsigned)(ET > (1ll << 20) ? (1ll << 20) : ET);
    k_edge3<<<gb, 256, 0, stream>>>(srcIdx, dstIdx, E, ET, ebuf, den, hA, hB);
    k_bias_elu<<<gsz(NF, 256), 256, 0, stream>>>(hB, bias, hb, NF);
  };

  // --- layer 1: h = x @ W1 (bf16 WMMA, K padded to 2624, W1 pre-transposed) ---
  k_cvt_bf16<<<gsz(N * Kp1, 256), 256, 0, stream>>>(x, xb, N, D, N, Kp1);
  k_cvt_bf16_T<<<gsz((long long)F * Kp1, 256), 256, 0, stream>>>(W1, w1t, D, F, Kp1);
  k_gemm_bf16<<<ggrid, 256, 0, stream>>>(xb, w1t, hA, nullptr, (int)N, Kp1);
  gat_layer(a1s, a1d, b1);

  // --- layer 2: h = elu1_out @ W2 ---
  k_cvt_bf16_T<<<gsz((long long)F * F, 256), 256, 0, stream>>>(W2, wt, F, F, F);
  k_gemm_bf16<<<ggrid, 256, 0, stream>>>(hb, wt, hA, nullptr, (int)N, F);
  gat_layer(a2s, a2d, b2);

  // --- linear1 + BN1 + elu ---
  k_cvt_bf16_T<<<gsz((long long)F * F, 256), 256, 0, stream>>>(lw1, wt, F, F, F);
  k_gemm_bf16<<<ggrid, 256, 0, stream>>>(hb, wt, hA, lb1, (int)N, F);
  k_bn_stats<<<(unsigned)F, 256, 0, stream>>>(hA, mu, var, (int)N);
  k_bn_apply<<<gsz(NF, 256), 256, 0, stream>>>(hA, mu, var, g1, be1, hB, hb, NF);

  // --- linear2 + BN2 + elu -> d_out ---
  k_cvt_bf16_T<<<gsz((long long)F * F, 256), 256, 0, stream>>>(lw2, wt, F, F, F);
  k_gemm_bf16<<<ggrid, 256, 0, stream>>>(hb, wt, hA, lb2, (int)N, F);
  k_bn_stats<<<(unsigned)F, 256, 0, stream>>>(hA, mu, var, (int)N);
  k_bn_apply<<<gsz(NF, 256), 256, 0, stream>>>(hA, mu, var, g2, be2,
                                               (float*)d_out, nullptr, NF);
}
